// res_pool_GCN_65309272703519
// MI455X (gfx1250) — compile-verified, matrix operand order checked by reading, not run on hardware
//
#include <hip/hip_runtime.h>
#include <math.h>

// ---------- types ----------
typedef __attribute__((ext_vector_type(16))) __bf16 v16bf;
typedef __attribute__((ext_vector_type(16))) float  v16f;
typedef __attribute__((ext_vector_type(8)))  float  v8f;

#define FIN   200
#define HDIM  128
#define GNUM  4096
#define NNODE 524288
#define HSTR  132   // LDS row stride in floats (avoid 128-stride bank conflicts)

// scalar fp32 -> bf16 RNE (used only in the tiny one-time pack kernel)
__device__ inline unsigned short f2bf(float f) {
    union { float f; unsigned u; } in; in.f = f;
    unsigned u = in.u;
    unsigned r = u + 0x7FFFu + ((u >> 16) & 1u);
    return (unsigned short)(r >> 16);
}

// vector fp32 -> bf16 via hardware converts (packed v_cvt on gfx1250)
__device__ inline v16bf cvt16(v16f f) {
    return __builtin_convertvector(f, v16bf);
}

// Build wave32 bf16 A-fragment (16x32 tile) from a row-major fp32 row pointer,
// no bounds checks (caller guarantees K in range). Per ISA layout:
// lanes 0-15 = row m, K in [k0, k0+8) and [k0+16, k0+24);
// lanes 16-31 = row m, K in [k0+8, k0+16) and [k0+24, k0+32).
__device__ inline v16bf loadA_fast(const float* __restrict__ rowp, int k0, int half) {
    int b1 = k0 + 8 * half;
    int b2 = b1 + 16;
    const float4* p1 = (const float4*)(rowp + b1);
    const float4* p2 = (const float4*)(rowp + b2);
    float4 q0 = p1[0], q1 = p1[1], q2 = p2[0], q3 = p2[1];
    v16f f;
    f[0]  = q0.x; f[1]  = q0.y; f[2]  = q0.z; f[3]  = q0.w;
    f[4]  = q1.x; f[5]  = q1.y; f[6]  = q1.z; f[7]  = q1.w;
    f[8]  = q2.x; f[9]  = q2.y; f[10] = q2.z; f[11] = q2.w;
    f[12] = q3.x; f[13] = q3.y; f[14] = q3.z; f[15] = q3.w;
    return cvt16(f);
}

// Tail A-fragment for layer-1 k-tile 6 (k0=192, K=200):
// half 0: K 192..199 valid in slots 0..7, rest zero. half 1: all zero.
__device__ inline v16bf loadA_tail(const float* __restrict__ rowp, int half) {
    v16f f;
#pragma unroll
    for (int i = 0; i < 16; ++i) f[i] = 0.0f;
    if (half == 0) {
        const float4* p1 = (const float4*)(rowp + 192);
        float4 q0 = p1[0], q1 = p1[1];
        f[0] = q0.x; f[1] = q0.y; f[2] = q0.z; f[3] = q0.w;
        f[4] = q1.x; f[5] = q1.y; f[6] = q1.z; f[7] = q1.w;
    }
    return cvt16(f);
}

// ---------- weight packing into B-fragment order ----------
// Packed layout: fragment (kt, nt) -> 32 lanes x 16 bf16, lane-major.
// lane l<16: col n=l, K in {0..7} U {16..23}; lane l+16: col n=l, K in {8..15} U {24..31}.
__global__ __launch_bounds__(256) void pack_kernel(
    const float* __restrict__ W1, const float* __restrict__ W2, const float* __restrict__ W3,
    unsigned short* __restrict__ W1p, unsigned short* __restrict__ W2p, unsigned short* __restrict__ W3p)
{
    int i = blockIdx.x * 256 + threadIdx.x;
    const float* W; unsigned short* dst; int K; int p;
    if (i < 28672)       { W = W1; dst = W1p; K = FIN;  p = i; }           // 7 ktiles * 4096
    else if (i < 45056)  { W = W2; dst = W2p; K = HDIM; p = i - 28672; }   // 4 ktiles
    else if (i < 61440)  { W = W3; dst = W3p; K = HDIM; p = i - 45056; }
    else return;
    int s    = p & 15;
    int lane = (p >> 4) & 31;
    int nt   = (p >> 9) & 7;
    int kt   = p >> 12;
    int half = lane >> 4;
    int n    = nt * 16 + (lane & 15);
    int v    = s >> 1, h2 = s & 1;
    int kk   = (v < 4 ? 2 * v + h2 : 16 + 2 * (v - 4) + h2) + 8 * half;
    int k    = kt * 32 + kk;
    float val = (k < K) ? W[k * HDIM + n] : 0.0f;
    dst[p] = f2bf(val);
}

__global__ __launch_bounds__(256) void zero_kernel(float* __restrict__ p, int n) {
    int i = blockIdx.x * 256 + threadIdx.x;
    if (i < n) p[i] = 0.0f;
}

// ---------- fused 3-layer MLP + per-layer segment-max pooling ----------
__device__ inline void bias_relu(v8f acc[8], const float* __restrict__ b, int ln) {
#pragma unroll
    for (int nt = 0; nt < 8; ++nt) {
        float bb = b[nt * 16 + ln];
#pragma unroll
        for (int r = 0; r < 8; ++r)
            acc[nt][r] = fmaxf(acc[nt][r] + bb, 0.0f);
    }
}

__device__ inline void store_acc(const v8f acc[8], float* base, int half, int ln) {
#pragma unroll
    for (int nt = 0; nt < 8; ++nt)
#pragma unroll
        for (int r = 0; r < 8; ++r)
            base[(r + 8 * half) * HSTR + nt * 16 + ln] = acc[nt][r];
}

// Column-per-thread pooling pass over the 64-row LDS tile; sorted segment ids
// mean ~1 atomic per distinct segment per column.
__device__ inline void pool_pass(const float* __restrict__ hbuf, const int* __restrict__ seg,
                                 int blockRow, float* __restrict__ pool, int c)
{
    int   cur = seg[blockRow];
    float m   = hbuf[c];
#pragma unroll 4
    for (int r = 1; r < 64; ++r) {
        int   s = seg[blockRow + r];
        float v = hbuf[r * HSTR + c];
        if (s != cur) {
            atomicMax((int*)(pool + (size_t)cur * HDIM + c), __float_as_int(m));
            cur = s; m = v;
        } else {
            m = fmaxf(m, v);
        }
    }
    atomicMax((int*)(pool + (size_t)cur * HDIM + c), __float_as_int(m));
}

__global__ __launch_bounds__(128) void fused_mlp_pool(
    const float* __restrict__ x, const int* __restrict__ seg,
    const unsigned short* __restrict__ W1p, const unsigned short* __restrict__ W2p,
    const unsigned short* __restrict__ W3p,
    const float* __restrict__ b1, const float* __restrict__ b2, const float* __restrict__ b3,
    float* __restrict__ pool1, float* __restrict__ pool2, float* __restrict__ pool3)
{
    __shared__ float hbuf[64 * HSTR];   // 33,792 B
    const int tid  = threadIdx.x;
    const int wave = tid >> 5;
    const int lane = tid & 31;
    const int half = lane >> 4;
    const int ln   = lane & 15;
    const int blockRow = blockIdx.x * 64;
    const int myrow    = blockRow + wave * 16 + ln;

    v8f acc[8];
#pragma unroll
    for (int nt = 0; nt < 8; ++nt)
#pragma unroll
        for (int r = 0; r < 8; ++r) acc[nt][r] = 0.0f;

    // ---- layer 1: [16x200] x [200x128]; 6 unguarded k-tiles + 1 tail ----
    const float* xrow = x + (size_t)myrow * FIN;
    for (int kt = 0; kt < 6; ++kt) {
        v16bf a = loadA_fast(xrow, kt * 32, half);
#pragma unroll
        for (int nt = 0; nt < 8; ++nt) {
            v16bf b = *(const v16bf*)(W1p + ((size_t)(kt * 8 + nt) * 512 + lane * 16));
            acc[nt] = __builtin_amdgcn_wmma_f32_16x16x32_bf16(
                false, a, false, b, (short)0, acc[nt], false, false);
        }
    }
    {
        v16bf a = loadA_tail(xrow, half);
#pragma unroll
        for (int nt = 0; nt < 8; ++nt) {
            v16bf b = *(const v16bf*)(W1p + ((size_t)(6 * 8 + nt) * 512 + lane * 16));
            acc[nt] = __builtin_amdgcn_wmma_f32_16x16x32_bf16(
                false, a, false, b, (short)0, acc[nt], false, false);
        }
    }
    bias_relu(acc, b1, ln);
    store_acc(acc, hbuf + (wave * 16) * HSTR, half, ln);
    __syncthreads();
    pool_pass(hbuf, seg, blockRow, pool1, tid);

    // ---- layer 2: A from LDS (fp32 h1 -> bf16) ----
    const float* hrow = hbuf + (size_t)(wave * 16 + ln) * HSTR;
#pragma unroll
    for (int nt = 0; nt < 8; ++nt)
#pragma unroll
        for (int r = 0; r < 8; ++r) acc[nt][r] = 0.0f;
    for (int kt = 0; kt < 4; ++kt) {
        v16bf a = loadA_fast(hrow, kt * 32, half);
#pragma unroll
        for (int nt = 0; nt < 8; ++nt) {
            v16bf b = *(const v16bf*)(W2p + ((size_t)(kt * 8 + nt) * 512 + lane * 16));
            acc[nt] = __builtin_amdgcn_wmma_f32_16x16x32_bf16(
                false, a, false, b, (short)0, acc[nt], false, false);
        }
    }
    bias_relu(acc, b2, ln);
    __syncthreads();                 // pooling-1 + all h1 A-reads done
    store_acc(acc, hbuf + (wave * 16) * HSTR, half, ln);
    __syncthreads();
    pool_pass(hbuf, seg, blockRow, pool2, tid);

    // ---- layer 3 ----
#pragma unroll
    for (int nt = 0; nt < 8; ++nt)
#pragma unroll
        for (int r = 0; r < 8; ++r) acc[nt][r] = 0.0f;
    for (int kt = 0; kt < 4; ++kt) {
        v16bf a = loadA_fast(hrow, kt * 32, half);
#pragma unroll
        for (int nt = 0; nt < 8; ++nt) {
            v16bf b = *(const v16bf*)(W3p + ((size_t)(kt * 8 + nt) * 512 + lane * 16));
            acc[nt] = __builtin_amdgcn_wmma_f32_16x16x32_bf16(
                false, a, false, b, (short)0, acc[nt], false, false);
        }
    }
    bias_relu(acc, b3, ln);
    __syncthreads();
    store_acc(acc, hbuf + (wave * 16) * HSTR, half, ln);
    __syncthreads();
    pool_pass(hbuf, seg, blockRow, pool3, tid);
}

// ---------- emb = p1+p2+p3 ; z = emb @ Wg ----------
__global__ __launch_bounds__(128) void emb_kernel(
    const float* __restrict__ p1, const float* __restrict__ p2, const float* __restrict__ p3,
    const float* __restrict__ Wg, float* __restrict__ emb, float* __restrict__ z)
{
    int g = blockIdx.x, c = threadIdx.x;
    size_t off = (size_t)g * HDIM + c;
    float e = p1[off] + p2[off] + p3[off];
    emb[off] = e;
    __shared__ float red[128];
    red[c] = e * Wg[c];
    __syncthreads();
    for (int s = 64; s > 0; s >>= 1) {
        if (c < s) red[c] += red[c + s];
        __syncthreads();
    }
    if (c == 0) z[g] = red[0];
}

// ---------- out = sigmoid(adj @ z + bg) ----------
__global__ __launch_bounds__(256) void gcn_kernel(
    const float* __restrict__ adj, const float* __restrict__ z,
    const float* __restrict__ bg, float* __restrict__ out)
{
    int g = blockIdx.x, t = threadIdx.x;
    float s = 0.0f;
    for (int j = t; j < GNUM; j += 256)
        s += adj[(size_t)g * GNUM + j] * z[j];
    __shared__ float red[256];
    red[t] = s;
    __syncthreads();
    for (int k = 128; k > 0; k >>= 1) {
        if (t < k) red[t] += red[t + k];
        __syncthreads();
    }
    if (t == 0) {
        float v = red[0] + bg[0];
        out[g] = 1.0f / (1.0f + expf(-v));
    }
}

// ---------- workspace layout (bytes) ----------
static const size_t OFF_P1  = 0;                       // 4096*128*4 = 2 MB
static const size_t OFF_P2  = 2097152;
static const size_t OFF_P3  = 4194304;
static const size_t OFF_Z   = 6291456;                 // 16 KB
static const size_t OFF_W1P = 6307840;                 // 224*128*2
static const size_t OFF_W2P = 6365184;                 // 128*128*2
static const size_t OFF_W3P = 6397952;                 // 128*128*2

extern "C" void kernel_launch(void* const* d_in, const int* in_sizes, int n_in,
                              void* d_out, int out_size, void* d_ws, size_t ws_size,
                              hipStream_t stream)
{
    const float* x   = (const float*)d_in[0];
    const int*   seg = (const int*)d_in[1];
    const float* adj = (const float*)d_in[2];
    const float* W1  = (const float*)d_in[3];
    const float* b1  = (const float*)d_in[4];
    const float* W2  = (const float*)d_in[5];
    const float* b2  = (const float*)d_in[6];
    const float* W3  = (const float*)d_in[7];
    const float* b3  = (const float*)d_in[8];
    const float* Wg  = (const float*)d_in[9];
    const float* bg  = (const float*)d_in[10];

    char* ws = (char*)d_ws;
    float* pool1 = (float*)(ws + OFF_P1);
    float* pool2 = (float*)(ws + OFF_P2);
    float* pool3 = (float*)(ws + OFF_P3);
    float* z     = (float*)(ws + OFF_Z);
    unsigned short* W1p = (unsigned short*)(ws + OFF_W1P);
    unsigned short* W2p = (unsigned short*)(ws + OFF_W2P);
    unsigned short* W3p = (unsigned short*)(ws + OFF_W3P);

    float* outv = (float*)d_out;       // [4096]
    float* emb  = outv + GNUM;         // [4096,128]

    // pools must be re-zeroed every call (atomicMax accumulates)
    int poolElems = GNUM * HDIM * 3;   // contiguous p1|p2|p3
    zero_kernel<<<(poolElems + 255) / 256, 256, 0, stream>>>(pool1, poolElems);

    pack_kernel<<<(61440 + 255) / 256, 256, 0, stream>>>(W1, W2, W3, W1p, W2p, W3p);

    fused_mlp_pool<<<NNODE / 64, 128, 0, stream>>>(
        x, seg, W1p, W2p, W3p, b1, b2, b3, pool1, pool2, pool3);

    emb_kernel<<<GNUM, 128, 0, stream>>>(pool1, pool2, pool3, Wg, emb, z);

    gcn_kernel<<<GNUM, 256, 0, stream>>>(adj, z, bg, outv);
}
